// rsdcnv4_73538430042724
// MI455X (gfx1250) — matrix operand order repaired and checked by reading, not compiled, not run on hardware
//
#include <hip/hip_runtime.h>
#include <hip/hip_bf16.h>
#include <math.h>

// ---------------------------------------------------------------------------
// DCNv4 residual block for MI455X (gfx1250, wave32).
// All GEMMs: v_wmma_f32_16x16x32_bf16 (bf16 in, f32 accumulate), K=128 fixed,
// fully unrolled, B pre-packed into the WMMA 32x16 fragment layout so the
// inner loop is pure b128 loads + wmma.
// ---------------------------------------------------------------------------

#define NB   4
#define CH   128
#define HH   64
#define WW   64
#define HWSZ 4096            // H*W
#define MROWS 16384          // N*H*W
#define GG   4
#define CG   32              // CH/GG
#define KK   9
#define OMW  108             // G*3*K total row width
#define KT   128             // GEMM reduction depth (all GEMMs)
#define NTPAD 8              // padded N-tiles for packed B (8*16 = 128 cols)

typedef __attribute__((ext_vector_type(16))) __bf16 v16bf;
typedef __attribute__((ext_vector_type(8)))  float  v8f;

union FragU { v16bf v; unsigned int u[8]; uint4 q[2]; };

__device__ __forceinline__ unsigned short f32_to_bf16(float f) {
  unsigned int u = __float_as_uint(f);
  unsigned int r = (u + 0x7FFFu + ((u >> 16) & 1u)) >> 16;
  return (unsigned short)r;
}

// ---------------------------------------------------------------------------
// NCHW f32 -> NHWC f32 (shortcut) + NHWC bf16 (GEMM operand)
// ---------------------------------------------------------------------------
__global__ void k_nchw_to_nhwc(const float* __restrict__ x,
                               float* __restrict__ xf,
                               unsigned short* __restrict__ xb) {
  int idx = blockIdx.x * blockDim.x + threadIdx.x;       // N*C*HW = 2097152
  if (idx >= NB * CH * HWSZ) return;
  int hw = idx & (HWSZ - 1);
  int c  = (idx >> 12) & (CH - 1);
  int n  = idx >> 19;
  float v = x[idx];                                      // coalesced read
  size_t o = ((size_t)((n << 12) + hw)) * CH + c;
  xf[o] = v;
  xb[o] = f32_to_bf16(v);
}

// ---------------------------------------------------------------------------
// Pack weight B (f32, KT x Nt, optionally transposed source) into the WMMA
// 32x16 B fragment (this file's k order, same as A): dst[((tileN*4 + kb32)*32 + lane)*8 + v] holds the
// bf16 pair {k0, k0+1} for column (tileN*16 + lane%16),
// k0 = kb32*32 + (lane/16)*16 + 2v.  Zero-filled for columns >= Nt.
// ---------------------------------------------------------------------------
__global__ void k_pack_B(const float* __restrict__ src,
                         unsigned int* __restrict__ dst,
                         int Nt, int trans) {
  int i = blockIdx.x * blockDim.x + threadIdx.x;         // NTPAD*4*32*8 = 8192
  if (i >= NTPAD * 4 * 32 * 8) return;
  int v     = i & 7;
  int lane  = (i >> 3) & 31;
  int kb32  = (i >> 8) & 3;
  int tileN = i >> 10;
  int half  = lane >> 4;
  int nl    = lane & 15;
  int ncol  = tileN * 16 + nl;
  int k0    = kb32 * 32 + half * 16 + v * 2;
  unsigned int lo = 0, hi = 0;
  if (ncol < Nt) {
    if (trans) {  // src is Nt x KT (e.g. Wc: B[k][n] = Wc[n][k])
      lo = f32_to_bf16(src[(size_t)ncol * KT + k0]);
      hi = f32_to_bf16(src[(size_t)ncol * KT + k0 + 1]);
    } else {      // src is KT x Nt row-major
      lo = f32_to_bf16(src[(size_t)k0 * Nt + ncol]);
      hi = f32_to_bf16(src[(size_t)(k0 + 1) * Nt + ncol]);
    }
  }
  dst[i] = lo | (hi << 16);
}

// ---------------------------------------------------------------------------
// bf16 WMMA GEMM: D(f32, M x Nt) = A(bf16, M x 128) * B + bias
// One wave computes a 16x32 D strip (two 16x16 tiles sharing the A fragment).
// K fully unrolled (4 steps of 32) -> 8 static v_wmma per wave.
// mode 0: D[row*Nt + col]   mode 1: D stored as NCHW f32 (final output)
// ---------------------------------------------------------------------------
__global__ void k_gemm_bf16_wmma(const unsigned short* __restrict__ A,
                                 const unsigned int* __restrict__ Bp,
                                 const float* __restrict__ bias,
                                 float* __restrict__ D,
                                 int M, int Nt, int numPairs, int mode) {
  int wid  = (blockIdx.x * blockDim.x + threadIdx.x) >> 5;
  int lane = threadIdx.x & 31;
  int totalTiles = (M >> 4) * numPairs;
  if (wid >= totalTiles) return;
  int pairN = wid % numPairs;
  int tileM = wid / numPairs;
  int mBase = tileM << 4;
  int half  = lane >> 4;
  int ml    = lane & 15;
  int tn0   = pairN * 2;
  int tn1   = tn0 + 1;

  v8f acc0 = {0.f, 0.f, 0.f, 0.f, 0.f, 0.f, 0.f, 0.f};
  v8f acc1 = {0.f, 0.f, 0.f, 0.f, 0.f, 0.f, 0.f, 0.f};

  const unsigned short* arow = A + (size_t)(mBase + ml) * KT;
  const unsigned int*   pb0  = Bp + ((size_t)tn0 * 4 * 32 + lane) * 8;
  const unsigned int*   pb1  = Bp + ((size_t)tn1 * 4 * 32 + lane) * 8;

#pragma unroll
  for (int kb32 = 0; kb32 < 4; ++kb32) {
    int kb = kb32 * 32;
    FragU a, b0, b1;
    // A fragment: K = 16*half .. 16*half+15 are 32 contiguous bytes -> two
    // b128 loads.
    a.q[0] = *reinterpret_cast<const uint4*>(arow + kb + half * 16);
    a.q[1] = *reinterpret_cast<const uint4*>(arow + kb + half * 16 + 8);
    // B fragments: pre-packed, 32 contiguous bytes per lane per K-step.
    b0.q[0] = *reinterpret_cast<const uint4*>(pb0 + kb32 * 256);
    b0.q[1] = *reinterpret_cast<const uint4*>(pb0 + kb32 * 256 + 4);
    b1.q[0] = *reinterpret_cast<const uint4*>(pb1 + kb32 * 256);
    b1.q[1] = *reinterpret_cast<const uint4*>(pb1 + kb32 * 256 + 4);
    acc0 = __builtin_amdgcn_wmma_f32_16x16x32_bf16(
        false, a.v, false, b0.v, (short)0, acc0, false, false);
    acc1 = __builtin_amdgcn_wmma_f32_16x16x32_bf16(
        false, a.v, false, b1.v, (short)0, acc1, false, false);
  }

#pragma unroll
  for (int t = 0; t < 2; ++t) {
    int ncol = (t ? tn1 : tn0) * 16 + ml;
    if (ncol >= Nt) continue;
    v8f& acc = t ? acc1 : acc0;
    float bv = bias ? bias[ncol] : 0.0f;
#pragma unroll
    for (int j = 0; j < 8; ++j) {
      int mrow = mBase + j + half * 8;   // D layout: lanes16-31 -> M=j+8
      float val = acc[j] + bv;
      if (mode == 0) {
        D[(size_t)mrow * Nt + ncol] = val;
      } else {  // NCHW store: row = n*HW + hw, col = output channel
        int ni = mrow >> 12;
        int hw = mrow & (HWSZ - 1);
        D[(((size_t)ni * CH + ncol) << 12) + hw] = val;
      }
    }
  }
}

// ---------------------------------------------------------------------------
// Deformable sampling: one wave per (n,h,w,g); lane = channel within group.
// ---------------------------------------------------------------------------
__global__ void k_dcnv4_sample(const float* __restrict__ value,
                               const float* __restrict__ om,
                               unsigned short* __restrict__ outb) {
  int wid  = (blockIdx.x * blockDim.x + threadIdx.x) >> 5;
  int lane = threadIdx.x & 31;
  if (wid >= NB * HWSZ * GG) return;
  int g  = wid & 3;
  int hw = (wid >> 2) & (HWSZ - 1);
  int n  = wid >> 14;
  int h  = hw >> 6;
  int w  = hw & 63;
  int row = (n << 12) + hw;

  const float* omp   = om + (size_t)row * OMW + g * 27;
  const float* vbase = value + ((size_t)(n << 12)) * CH + g * CG + lane;

  float acc = 0.f;
#pragma unroll
  for (int k = 0; k < KK; ++k) {
    float offx = omp[2 * k];
    float offy = omp[2 * k + 1];
    float msk  = omp[18 + k];
    float ly = (float)(h + (k / 3) - 1) + offy;
    float lx = (float)(w + (k % 3) - 1) + offx;
    float y0f = floorf(ly), x0f = floorf(lx);
    float wy = ly - y0f, wx = lx - x0f;
    int y0 = (int)y0f, x0 = (int)x0f;
    float s = 0.f;
#pragma unroll
    for (int cy = 0; cy < 2; ++cy) {
#pragma unroll
      for (int cx = 0; cx < 2; ++cx) {
        int yi = y0 + cy, xi = x0 + cx;
        float wgt = (cy ? wy : 1.f - wy) * (cx ? wx : 1.f - wx);
        bool valid = (yi >= 0) & (yi < HH) & (xi >= 0) & (xi < WW);
        int yc = min(max(yi, 0), HH - 1);
        int xc = min(max(xi, 0), WW - 1);
        float v = vbase[((size_t)((yc << 6) + xc)) * CH];  // 128B coalesced
        s += valid ? v * wgt : 0.f;
      }
    }
    acc += msk * s;
  }
  outb[(size_t)row * CH + g * CG + lane] = f32_to_bf16(acc);
}

// ---------------------------------------------------------------------------
// BatchNorm: deterministic two-stage reduction over M rows per channel.
// ---------------------------------------------------------------------------
#define RBLK 64                // partial-sum blocks (16384/64 = 256 rows each)

__global__ void k_bn_reduce(const float* __restrict__ x,
                            float* __restrict__ partials) {
  int c = threadIdx.x;         // 0..127, coalesced across channels
  int b = blockIdx.x;          // 0..RBLK-1
  const float* p = x + (size_t)(b * (MROWS / RBLK)) * CH + c;
  float s = 0.f, s2 = 0.f;
  for (int r = 0; r < MROWS / RBLK; ++r) {
    float v = p[(size_t)r * CH];
    s += v; s2 += v * v;
  }
  partials[b * CH + c] = s;
  partials[RBLK * CH + b * CH + c] = s2;
}

__global__ void k_bn_finalize(const float* __restrict__ partials,
                              float* __restrict__ stats) {
  int c = threadIdx.x;
  float s = 0.f, s2 = 0.f;
  for (int b = 0; b < RBLK; ++b) {          // fixed order -> deterministic
    s  += partials[b * CH + c];
    s2 += partials[RBLK * CH + b * CH + c];
  }
  float mean = s * (1.f / MROWS);
  float var  = s2 * (1.f / MROWS) - mean * mean;
  stats[c]      = mean;
  stats[CH + c] = rsqrtf(var + 1e-5f);
}

__global__ void k_bn_apply(const float* __restrict__ x,
                           const float* __restrict__ stats,
                           const float* __restrict__ gamma,
                           const float* __restrict__ beta,
                           const float* __restrict__ addsrc,
                           int doRelu,
                           float* __restrict__ yf,
                           unsigned short* __restrict__ yb) {
  int idx = blockIdx.x * blockDim.x + threadIdx.x;       // < MROWS*CH
  if (idx >= MROWS * CH) return;
  int c = idx & (CH - 1);
  float v = (x[idx] - stats[c]) * stats[CH + c] * gamma[c] + beta[c];
  if (doRelu) v = fmaxf(v, 0.f);
  if (addsrc) v += addsrc[idx];
  yf[idx] = v;
  yb[idx] = f32_to_bf16(v);
}

// ---------------------------------------------------------------------------
// Host orchestration
// ---------------------------------------------------------------------------
static inline size_t align256(size_t x) { return (x + 255) & ~(size_t)255; }

extern "C" void kernel_launch(void* const* d_in, const int* in_sizes, int n_in,
                              void* d_out, int out_size, void* d_ws, size_t ws_size,
                              hipStream_t stream) {
  (void)in_sizes; (void)n_in; (void)out_size; (void)ws_size;

  const float* x     = (const float*)d_in[0];
  const float* Wv    = (const float*)d_in[1];
  const float* bv    = (const float*)d_in[2];
  const float* Wom   = (const float*)d_in[3];
  const float* bom   = (const float*)d_in[4];
  const float* Wo    = (const float*)d_in[5];
  const float* gamma = (const float*)d_in[6];
  const float* beta  = (const float*)d_in[7];
  const float* Wc    = (const float*)d_in[8];
  float* out = (float*)d_out;

  // Workspace carve (all fits comfortably in the 192MB L2)
  char* p = (char*)d_ws;
  float*          xnhwc = (float*)p;          p += align256((size_t)MROWS * CH * 4);
  unsigned short* tb    = (unsigned short*)p; p += align256((size_t)MROWS * CH * 2);
  float*          value = (float*)p;          p += align256((size_t)MROWS * CH * 4);
  float*          omb   = (float*)p;          p += align256((size_t)MROWS * OMW * 4);
  unsigned short* samp  = (unsigned short*)p; p += align256((size_t)MROWS * CH * 2);
  float*          zf    = (float*)p;          p += align256((size_t)MROWS * CH * 4);
  float*          part  = (float*)p;          p += align256((size_t)2 * RBLK * CH * 4);
  float*          stats = (float*)p;          p += align256((size_t)2 * CH * 4);
  const size_t PACKB = (size_t)NTPAD * 4 * 32 * 8;       // uints per packed B
  unsigned int*   wvP   = (unsigned int*)p;   p += align256(PACKB * 4);
  unsigned int*   womP  = (unsigned int*)p;   p += align256(PACKB * 4);
  unsigned int*   woP   = (unsigned int*)p;   p += align256(PACKB * 4);
  unsigned int*   wcP   = (unsigned int*)p;   p += align256(PACKB * 4);

  // --- pack weights into WMMA B-fragment layout (bf16 pairs, zero-padded)
  const int packBlk = ((int)PACKB + 255) / 256;          // 32 blocks
  k_pack_B<<<packBlk, 256, 0, stream>>>(Wv,  wvP,  CH,  0);
  k_pack_B<<<packBlk, 256, 0, stream>>>(Wom, womP, OMW, 0);
  k_pack_B<<<packBlk, 256, 0, stream>>>(Wo,  woP,  CH,  0);
  k_pack_B<<<packBlk, 256, 0, stream>>>(Wc,  wcP,  CH,  1);  // transposed read

  // --- layout: NCHW -> NHWC (f32 shortcut + bf16 GEMM operand)
  k_nchw_to_nhwc<<<(MROWS * CH + 255) / 256, 256, 0, stream>>>(x, xnhwc, tb);

  const int numPairs = NTPAD / 2;                        // 4 (16x32 per wave)
  const int gemmBlk  = ((MROWS / 16) * numPairs * 32 + 255) / 256;  // 512
  const int sampBlk  = (NB * HWSZ * GG * 32 + 255) / 256;
  const int eltBlk   = (MROWS * CH + 255) / 256;

  for (int pass = 0; pass < 2; ++pass) {
    // value = t @ Wv + bv        (16384x128x128, WMMA bf16)
    k_gemm_bf16_wmma<<<gemmBlk, 256, 0, stream>>>(tb, wvP, bv, value,
                                                  MROWS, CH, numPairs, 0);
    // om = t @ Wom + bom         (16384x108x128, WMMA bf16, edge-guarded)
    k_gemm_bf16_wmma<<<gemmBlk, 256, 0, stream>>>(tb, womP, bom, omb,
                                                  MROWS, OMW, numPairs, 0);
    // deformable bilinear gather + mask-weighted sum -> bf16
    k_dcnv4_sample<<<sampBlk, 256, 0, stream>>>(value, omb, samp);
    // z = sampled @ Wo           (16384x128x128, WMMA bf16)
    k_gemm_bf16_wmma<<<gemmBlk, 256, 0, stream>>>(samp, woP, nullptr, zf,
                                                  MROWS, CH, numPairs, 0);
    // BatchNorm (deterministic two-stage) + ReLU (pass 0) / +shortcut (pass 1)
    k_bn_reduce  <<<RBLK, CH, 0, stream>>>(zf, part);
    k_bn_finalize<<<1,    CH, 0, stream>>>(part, stats);
    k_bn_apply<<<eltBlk, 256, 0, stream>>>(zf, stats, gamma, beta,
                                           pass == 0 ? nullptr : xnhwc,
                                           pass == 0 ? 1 : 0,
                                           zf, tb);
  }

  // final: out[n,o,h,w] = sum_c y[n,c,h,w] * Wc[o,c]  (WMMA, NCHW store)
  k_gemm_bf16_wmma<<<gemmBlk, 256, 0, stream>>>(tb, wcP, nullptr, out,
                                                MROWS, CH, numPairs, 1);
}
